// CharRNN_18906446037470
// MI455X (gfx1250) — matrix operand order repaired and checked by reading, not compiled, not run on hardware
//
#include <hip/hip_runtime.h>

// ---------------------------------------------------------------------------
// BN-LSTM CharRNN for MI455X (gfx1250, wave32, WMMA).
// GEMMs: bf16 inputs, fp32 accumulate via v_wmma_f32_16x16x32_bf16, with
// double-buffered LDS staging so global latency overlaps the matrix pipe.
// ---------------------------------------------------------------------------

typedef __bf16 bf16;
typedef __attribute__((ext_vector_type(16))) __bf16 v16bf;
typedef __attribute__((ext_vector_type(8)))  __bf16 v8bf;
typedef __attribute__((ext_vector_type(8)))  float  v8f;

#define BB 256      // batch
#define TT 64       // seq len
#define EE 512      // embed dim
#define HH 2048     // hidden
#define H4 (4*HH)   // gate dim
#define PP 512      // projection

#define BLK_M 64
#define BLK_N 256
#define BLK_K 32
#define LDS_LD 40   // padded LDS row stride (32 data + 8 pad bf16) -> 80B rows

union Frag16 { v16bf v; v8bf h[2]; };

// ---------------------------------------------------------------------------
// C[M,N] (f32) = A[M,K] (bf16 row-major) * Bt[N,K] (bf16 N-major).
// Block: 256 threads = 8 waves as 2(M) x 4(N); block tile 64x256; wave tile
// 32x64 (2x4 WMMA accumulators). K pipelined in 32-wide chunks through
// double-buffered LDS; chunk k+1's global loads are issued before chunk k's
// WMMAs so loadcnt waits retire behind compute.
// Requires M%64==0, N%256==0, K%32==0.
// ---------------------------------------------------------------------------
__global__ __launch_bounds__(256) void gemm_bf16_wmma(
    const bf16* __restrict__ A, const bf16* __restrict__ Bt,
    float* __restrict__ C, int M, int N, int K,
    const float* __restrict__ bias, int swizB, int swizT)
{
    __shared__ __align__(16) bf16 As[2][BLK_M * LDS_LD];
    __shared__ __align__(16) bf16 Bs[2][BLK_N * LDS_LD];

    const int tid   = threadIdx.x;
    const int lane  = tid & 31;
    const int wave  = tid >> 5;
    const int waveM = wave >> 2;                 // 0..1
    const int waveN = wave & 3;                  // 0..3
    const int half  = lane >> 4;
    const int l15   = lane & 15;
    const int rowBlock = blockIdx.y * BLK_M;
    const int colBlock = blockIdx.x * BLK_N;

    // Cooperative copy assignments: 64 A-rows x 4 chunks, 4x(64 B-cols) x 4 chunks.
    const int cRow   = tid >> 2;                 // 0..63
    const int cChunk = tid & 3;                  // 0..3 (8 bf16 each)
    const bf16* __restrict__ Ag = A  + (size_t)(rowBlock + cRow) * K + cChunk * 8;
    const bf16* __restrict__ Bg = Bt + (size_t)(colBlock + cRow) * K + cChunk * 8;

    v8bf aReg;
    v8bf bReg[4];
    auto gload = [&](int k0) {
        aReg = *(const v8bf*)(Ag + k0);
        #pragma unroll
        for (int i = 0; i < 4; ++i)
            bReg[i] = *(const v8bf*)(Bg + (size_t)(i * 64) * K + k0);
    };
    auto sstore = [&](int buf) {
        *(v8bf*)&As[buf][cRow * LDS_LD + cChunk * 8] = aReg;
        #pragma unroll
        for (int i = 0; i < 4; ++i)
            *(v8bf*)&Bs[buf][(i * 64 + cRow) * LDS_LD + cChunk * 8] = bReg[i];
    };

    v8f acc[2][4];
    #pragma unroll
    for (int m = 0; m < 2; ++m)
        #pragma unroll
        for (int n = 0; n < 4; ++n) acc[m][n] = (v8f){};

    const int nK = K / BLK_K;
    gload(0);
    for (int kc = 0; kc < nK; ++kc) {
        const int buf = kc & 1;
        sstore(buf);
        __syncthreads();
        if (kc + 1 < nK) gload((kc + 1) * BLK_K);   // overlap with WMMAs below

        // A fragments: lane l15 -> row, half selects K {0..7,16..23}/{8..15,24..31}
        Frag16 afr[2];
        #pragma unroll
        for (int m = 0; m < 2; ++m) {
            const bf16* pa = &As[buf][(waveM * 32 + m * 16 + l15) * LDS_LD + half * 8];
            afr[m].h[0] = *(const v8bf*)pa;
            afr[m].h[1] = *(const v8bf*)(pa + 16);
        }
        // B fragments: lane l15 -> column, half selects K {0..7,16..23}/{8..15,24..31}
        #pragma unroll
        for (int n = 0; n < 4; ++n) {
            const bf16* pb = &Bs[buf][(waveN * 64 + n * 16 + l15) * LDS_LD + half * 8];
            Frag16 bfr;
            bfr.h[0] = *(const v8bf*)pb;
            bfr.h[1] = *(const v8bf*)(pb + 16);
            #pragma unroll
            for (int m = 0; m < 2; ++m)
                acc[m][n] = __builtin_amdgcn_wmma_f32_16x16x32_bf16(
                    false, afr[m].v, false, bfr.v, (short)0, acc[m][n], false, false);
        }
        __syncthreads();
    }

    // D layout: VGPR r -> row half*8+r, column l15.
    #pragma unroll
    for (int m = 0; m < 2; ++m) {
        #pragma unroll
        for (int n = 0; n < 4; ++n) {
            #pragma unroll
            for (int r = 0; r < 8; ++r) {
                int row = rowBlock + waveM * 32 + m * 16 + half * 8 + r;
                int col = colBlock + waveN * 64 + n * 16 + l15;
                float v = acc[m][n][r];
                if (bias) v += bias[col];
                int orow = row;
                if (swizT > 0) {                   // rows ordered t*B+b -> b*T+t
                    int t = row / swizB;
                    int b = row - t * swizB;
                    orow = b * swizT + t;
                }
                C[(size_t)orow * N + col] = v;
            }
        }
    }
}

// ---------------------------------------------------------------------------
// Per-column batch stats (mean, rsqrt(var+eps)) for TWO arrays in one pass.
// One thread per column; coalesced row sweeps.
// ---------------------------------------------------------------------------
__global__ void col_stats2(const float* __restrict__ X0, const float* __restrict__ X1,
                           int rows, int cols,
                           float* __restrict__ m0, float* __restrict__ i0,
                           float* __restrict__ m1, float* __restrict__ i1)
{
    int c = blockIdx.x * blockDim.x + threadIdx.x;
    if (c >= cols) return;
    float s0 = 0.f, q0 = 0.f, s1 = 0.f, q1 = 0.f;
    for (int r = 0; r < rows; ++r) {
        float a = X0[(size_t)r * cols + c];
        float b = X1[(size_t)r * cols + c];
        s0 += a; q0 += a * a;
        s1 += b; q1 += b * b;
    }
    float ma = s0 / rows, mb = s1 / rows;
    m0[c] = ma; i0[c] = rsqrtf(q0 / rows - ma * ma + 1e-5f);
    m1[c] = mb; i1[c] = rsqrtf(q1 / rows - mb * mb + 1e-5f);
}

__global__ void col_stats(const float* __restrict__ X, int rows, int cols,
                          float* __restrict__ mean, float* __restrict__ inv)
{
    int c = blockIdx.x * blockDim.x + threadIdx.x;
    if (c >= cols) return;
    float s = 0.f, s2 = 0.f;
    for (int r = 0; r < rows; ++r) {
        float v = X[(size_t)r * cols + c];
        s += v; s2 += v * v;
    }
    float m = s / rows;
    mean[c] = m;
    inv[c]  = rsqrtf(s2 / rows - m * m + 1e-5f);
}

__device__ __forceinline__ float sigmoidf(float x) { return 1.f / (1.f + __expf(-x)); }

// gates = bn(xg)*gx + bn(hg)*gh + b ; cell update in place; cache sigmoid(o).
__global__ void gate_fuse(const float* __restrict__ xg, const float* __restrict__ hg,
                          const float* __restrict__ mx, const float* __restrict__ ix,
                          const float* __restrict__ mh, const float* __restrict__ ih,
                          const float* __restrict__ gx, const float* __restrict__ gh,
                          const float* __restrict__ bvec,
                          float* __restrict__ c, float* __restrict__ osig)
{
    int idx = blockIdx.x * blockDim.x + threadIdx.x;   // over B*H
    int b = idx / HH, j = idx - b * HH;
    size_t rb = (size_t)b * H4;
    auto gate = [&](int col) -> float {
        return (xg[rb + col] - mx[col]) * ix[col] * gx[col]
             + (hg[rb + col] - mh[col]) * ih[col] * gh[col]
             + bvec[col];
    };
    float gi = gate(j);
    float gj = gate(HH + j);
    float gf = gate(2 * HH + j);
    float go = gate(3 * HH + j);
    float c2 = sigmoidf(gf + 1.f) * c[idx] + sigmoidf(gi) * tanhf(gj);
    c[idx]    = c2;
    osig[idx] = sigmoidf(go);
}

// h = sigmoid(o) * tanh(bn(c)*gc + bc); emit bf16 into the [T,B,H] history.
__global__ void h_out(const float* __restrict__ c, const float* __restrict__ osig,
                      const float* __restrict__ mc, const float* __restrict__ ic,
                      const float* __restrict__ gc, const float* __restrict__ bc,
                      bf16* __restrict__ hall_t)
{
    int idx = blockIdx.x * blockDim.x + threadIdx.x;   // over B*H
    int j = idx & (HH - 1);
    float hn = osig[idx] * tanhf(gc[j] * (c[idx] - mc[j]) * ic[j] + bc[j]);
    hall_t[idx] = (bf16)hn;
}

// Embedding gather -> bf16, laid out [T, B, E] so step slices are contiguous.
__global__ void embed_gather(const int* __restrict__ ids,
                             const float* __restrict__ emb,
                             bf16* __restrict__ x)
{
    int flat = blockIdx.x;          // b*T + t
    int b = flat / TT, t = flat - b * TT;
    const float* src = emb + (size_t)ids[flat] * EE;
    bf16* dst = x + ((size_t)t * BB + b) * EE;
    for (int e = threadIdx.x; e < EE; e += blockDim.x)
        dst[e] = (bf16)src[e];
}

// W[K,N] f32 row-major -> Wt[N,K] bf16 (N-major) for the WMMA B path.
__global__ void transpose_to_bf16(const float* __restrict__ W,
                                  bf16* __restrict__ Wt, int K, int N)
{
    size_t i = (size_t)blockIdx.x * blockDim.x + threadIdx.x;
    if (i >= (size_t)K * N) return;
    int k = (int)(i / N), n = (int)(i - (size_t)k * N);
    Wt[(size_t)n * K + k] = (bf16)W[i];
}

__global__ void fill_f32(float* p, size_t n)
{
    size_t i = (size_t)blockIdx.x * blockDim.x + threadIdx.x;
    if (i < n) p[i] = 0.f;
}
__global__ void fill_bf16(bf16* p, size_t n)
{
    size_t i = (size_t)blockIdx.x * blockDim.x + threadIdx.x;
    if (i < n) p[i] = (bf16)0.f;
}

// ---------------------------------------------------------------------------
extern "C" void kernel_launch(void* const* d_in, const int* in_sizes, int n_in,
                              void* d_out, int out_size, void* d_ws, size_t ws_size,
                              hipStream_t stream)
{
    (void)in_sizes; (void)n_in; (void)out_size; (void)ws_size;

    const int*   ids = (const int*)  d_in[0];
    const float* emb = (const float*)d_in[1];
    const float* Wx1 = (const float*)d_in[2];
    const float* Wh1 = (const float*)d_in[3];
    const float* b1  = (const float*)d_in[4];
    const float* gx1 = (const float*)d_in[5];
    const float* gh1 = (const float*)d_in[6];
    const float* gc1 = (const float*)d_in[7];
    const float* bc1 = (const float*)d_in[8];
    const float* Wx2 = (const float*)d_in[9];
    const float* Wh2 = (const float*)d_in[10];
    const float* b2  = (const float*)d_in[11];
    const float* gx2 = (const float*)d_in[12];
    const float* gh2 = (const float*)d_in[13];
    const float* gc2 = (const float*)d_in[14];
    const float* bc2 = (const float*)d_in[15];
    const float* Wp  = (const float*)d_in[16];
    const float* bp  = (const float*)d_in[17];
    float* out = (float*)d_out;

    // ---- workspace carve-out (256B aligned slices) ----
    char* p = (char*)d_ws;
    auto alloc = [&](size_t bytes) -> char* {
        char* r = p;
        p += (bytes + 255) & ~(size_t)255;
        return r;
    };
    bf16* Wx1t = (bf16*)alloc((size_t)EE * H4 * 2);   // [4H, E]
    bf16* Wh1t = (bf16*)alloc((size_t)HH * H4 * 2);   // [4H, H]
    bf16* Wx2t = (bf16*)alloc((size_t)HH * H4 * 2);
    bf16* Wh2t = (bf16*)alloc((size_t)HH * H4 * 2);
    bf16* Wpt  = (bf16*)alloc((size_t)HH * PP * 2);   // [P, H]
    bf16* x1    = (bf16*)alloc((size_t)TT * BB * EE * 2);  // [T,B,E]
    bf16* h1all = (bf16*)alloc((size_t)TT * BB * HH * 2);  // [T,B,H]
    bf16* h2all = (bf16*)alloc((size_t)TT * BB * HH * 2);
    bf16* zeroh = (bf16*)alloc((size_t)BB * HH * 2);       // h_{-1} = 0
    float* xg   = (float*)alloc((size_t)BB * H4 * 4);
    float* hg   = (float*)alloc((size_t)BB * H4 * 4);
    float* cbuf = (float*)alloc((size_t)BB * HH * 4);
    float* osig = (float*)alloc((size_t)BB * HH * 4);
    float* mx = (float*)alloc(H4 * 4);
    float* ix = (float*)alloc(H4 * 4);
    float* mh = (float*)alloc(H4 * 4);
    float* ih = (float*)alloc(H4 * 4);
    float* mc = (float*)alloc(HH * 4);
    float* ic = (float*)alloc(HH * 4);

    // ---- one-time prep: weight transpose/convert + embedding gather ----
    auto tgrid = [](size_t n) { return (unsigned)((n + 255) / 256); };
    transpose_to_bf16<<<tgrid((size_t)EE * H4), 256, 0, stream>>>(Wx1, Wx1t, EE, H4);
    transpose_to_bf16<<<tgrid((size_t)HH * H4), 256, 0, stream>>>(Wh1, Wh1t, HH, H4);
    transpose_to_bf16<<<tgrid((size_t)HH * H4), 256, 0, stream>>>(Wx2, Wx2t, HH, H4);
    transpose_to_bf16<<<tgrid((size_t)HH * H4), 256, 0, stream>>>(Wh2, Wh2t, HH, H4);
    transpose_to_bf16<<<tgrid((size_t)HH * PP), 256, 0, stream>>>(Wp,  Wpt,  HH, PP);
    embed_gather<<<BB * TT, 128, 0, stream>>>(ids, emb, x1);
    fill_bf16<<<tgrid((size_t)BB * HH), 256, 0, stream>>>(zeroh, (size_t)BB * HH);

    const dim3 gGates(H4 / BLK_N, BB / BLK_M);   // GEMM grid for [B,4H] outputs
    const int  nBH = BB * HH;

    auto run_layer = [&](const bf16* xin, int Din,
                         const bf16* Wxt, const bf16* Wht,
                         const float* bvec, const float* gxv, const float* ghv,
                         const float* gcv, const float* bcv, bf16* hall) {
        fill_f32<<<tgrid(nBH), 256, 0, stream>>>(cbuf, (size_t)nBH);
        for (int t = 0; t < TT; ++t) {
            const bf16* At = xin + (size_t)t * BB * Din;
            gemm_bf16_wmma<<<gGates, 256, 0, stream>>>(At, Wxt, xg, BB, H4, Din,
                                                       nullptr, 0, 0);
            const bf16* Ah = (t == 0) ? zeroh : hall + (size_t)(t - 1) * BB * HH;
            gemm_bf16_wmma<<<gGates, 256, 0, stream>>>(Ah, Wht, hg, BB, H4, HH,
                                                       nullptr, 0, 0);
            col_stats2<<<H4 / 256, 256, 0, stream>>>(xg, hg, BB, H4, mx, ix, mh, ih);
            gate_fuse<<<nBH / 256, 256, 0, stream>>>(xg, hg, mx, ix, mh, ih,
                                                     gxv, ghv, bvec, cbuf, osig);
            col_stats<<<HH / 256, 256, 0, stream>>>(cbuf, BB, HH, mc, ic);
            h_out<<<nBH / 256, 256, 0, stream>>>(cbuf, osig, mc, ic, gcv, bcv,
                                                 hall + (size_t)t * BB * HH);
        }
    };

    run_layer(x1,    EE, Wx1t, Wh1t, b1, gx1, gh1, gc1, bc1, h1all);
    run_layer(h1all, HH, Wx2t, Wh2t, b2, gx2, gh2, gc2, bc2, h2all);

    // Output projection: rows come in [T,B] order -> swizzle stores to [B*T, P]
    gemm_bf16_wmma<<<dim3(PP / BLK_N, (BB * TT) / BLK_M), 256, 0, stream>>>(
        h2all, Wpt, out, BB * TT, PP, HH, bp, BB, TT);
}
